// HAGMoE_81243601371221
// MI455X (gfx1250) — compile-verified, matrix operand order checked by reading, not run on hardware
//
#include <hip/hip_runtime.h>
#include <math.h>

typedef __bf16 bf16;
typedef __attribute__((ext_vector_type(16))) __bf16 v16bf;
typedef __attribute__((ext_vector_type(8)))  __bf16 v8bf;
typedef __attribute__((ext_vector_type(8)))  float  v8f;

#define N_TOK 2048
#define HDIM  1024
#define IDIM  4096
#define NG    3
#define NE    8

constexpr int BM = 128, BN = 128, BK = 32;
constexpr int LDT = BK + 8;   // bf16 elems; 80B row stride (16B aligned, bank-spread)

// ---------------------------------------------------------------- helpers
__device__ __forceinline__ void async_b128(unsigned lds_addr, unsigned gvoff,
                                           const void* gbase) {
    // GVS mode: SGPR 64-bit base + 32-bit VGPR byte offset; LDS dest addr in VGPR.
    asm volatile("global_load_async_to_lds_b128 %0, %1, %2 offset:0"
                 :: "v"(lds_addr), "v"(gvoff), "s"(gbase) : "memory");
}

// ---------------------------------------------------------------- out = x
__global__ __launch_bounds__(256) void init_out_kernel(const float* __restrict__ x,
                                                       float* __restrict__ out) {
    int i = blockIdx.x * 256 + threadIdx.x;
    ((float4*)out)[i] = ((const float4*)x)[i];
}

// xb = bf16(x) [N,H];  xcb = bf16([x|cond]) [N,2H]
__global__ __launch_bounds__(256) void prep_x_kernel(const float* __restrict__ x,
                                                     const float* __restrict__ cond,
                                                     bf16* __restrict__ xb,
                                                     bf16* __restrict__ xcb) {
    int i = blockIdx.x * 256 + threadIdx.x;       // over N*H/4 float4 groups
    int row = i >> 8;                             // H/4 = 256 groups per row
    int c4  = (i & 255) * 4;
    float4 vx = ((const float4*)x)[i];
    float4 vc = ((const float4*)cond)[i];
    bf16 tx[4] = {(bf16)vx.x, (bf16)vx.y, (bf16)vx.z, (bf16)vx.w};
    bf16 tc[4] = {(bf16)vc.x, (bf16)vc.y, (bf16)vc.z, (bf16)vc.w};
    *(uint2*)&xb [(size_t)row * HDIM + c4]             = *(uint2*)tx;
    *(uint2*)&xcb[(size_t)row * 2 * HDIM + c4]         = *(uint2*)tx;
    *(uint2*)&xcb[(size_t)row * 2 * HDIM + HDIM + c4]  = *(uint2*)tc;
}

// in fp32 [R][C] -> out bf16 [C][R]  (32x32 LDS tile transpose + convert)
__global__ __launch_bounds__(256) void transpose_conv_kernel(const float* __restrict__ in,
                                                             bf16* __restrict__ out,
                                                             int R, int C) {
    __shared__ bf16 tile[32][33];
    int bc = blockIdx.x * 32, br = blockIdx.y * 32;
    int t = threadIdx.x;
    int lr = t >> 3, lc4 = (t & 7) * 4;
    float4 v = *(const float4*)(in + (size_t)(br + lr) * C + bc + lc4);
    tile[lr][lc4 + 0] = (bf16)v.x; tile[lr][lc4 + 1] = (bf16)v.y;
    tile[lr][lc4 + 2] = (bf16)v.z; tile[lr][lc4 + 3] = (bf16)v.w;
    __syncthreads();
    int wc = t >> 3, wr4 = (t & 7) * 4;
    bf16 o[4] = {tile[wr4 + 0][wc], tile[wr4 + 1][wc],
                 tile[wr4 + 2][wc], tile[wr4 + 3][wc]};
    *(uint2*)&out[(size_t)(bc + wc) * R + br + wr4] = *(uint2*)o;
}

// ------------------------------------------------------------- WMMA GEMM
// A bf16 [M][K] row-major, B bf16 [Ncols][K] row-major (pre-transposed).
// MODE 0: C = A@B^T + bias           -> fp32 store (z)
// MODE 1: C = gelu(A@B^T + bias)     -> bf16 store (hbuf)
// MODE 2: out += w[row]*(A@B^T+bias) -> fp32 accumulate
template <int MODE>
__global__ __launch_bounds__(256) void gemm_bf16_async_kernel(
    const bf16* __restrict__ A, const bf16* __restrict__ B,
    const float* __restrict__ bias, void* __restrict__ Cptr,
    const float* __restrict__ wrow, int K, int Ncols) {

    __shared__ bf16 As[2][BM * LDT];
    __shared__ bf16 Bs[2][BN * LDT];

    const int tid  = threadIdx.x;
    const int bM   = blockIdx.y * BM;
    const int bN   = blockIdx.x * BN;
    const int lane = tid & 31;
    const int wid  = tid >> 5;
    const int wm   = (wid & 1) * 64;
    const int wn   = (wid >> 1) * 32;
    const int half = lane >> 4;
    const int lrow = lane & 15;

    // per-thread staging coordinates (two b128 each for A and B)
    const int r0  = tid >> 2;              // 0..63
    const int r1  = r0 + 64;               // 64..127
    const int c8  = (tid & 3) * 8;         // 0,8,16,24
    const unsigned aoff0 = (unsigned)(((size_t)(bM + r0) * K + c8) * 2);
    const unsigned aoff1 = (unsigned)(((size_t)(bM + r1) * K + c8) * 2);
    const unsigned boff0 = (unsigned)(((size_t)(bN + r0) * K + c8) * 2);
    const unsigned boff1 = (unsigned)(((size_t)(bN + r1) * K + c8) * 2);

    v8f acc[4][2] = {};

    auto stage = [&](int kt, int buf) {
        unsigned kb = (unsigned)(kt * 2);
        async_b128((unsigned)(uintptr_t)&As[buf][r0 * LDT + c8], aoff0 + kb, A);
        async_b128((unsigned)(uintptr_t)&As[buf][r1 * LDT + c8], aoff1 + kb, A);
        async_b128((unsigned)(uintptr_t)&Bs[buf][r0 * LDT + c8], boff0 + kb, B);
        async_b128((unsigned)(uintptr_t)&Bs[buf][r1 * LDT + c8], boff1 + kb, B);
    };

    const int nk = K / BK;
    stage(0, 0);
    for (int i = 0; i < nk; ++i) {
        const int cur = i & 1;
        if (i + 1 < nk) {
            stage((i + 1) * BK, cur ^ 1);
            asm volatile("s_wait_asynccnt 4" ::: "memory");   // tile i landed
        } else {
            asm volatile("s_wait_asynccnt 0" ::: "memory");
        }
        __syncthreads();

        v16bf af[4], bfr[2];
#pragma unroll
        for (int q = 0; q < 4; ++q) {
            const bf16* p = &As[cur][(wm + q * 16 + lrow) * LDT + half * 8];
            v8bf lo = *(const v8bf*)p;
            v8bf hi = *(const v8bf*)(p + 16);
#pragma unroll
            for (int e = 0; e < 8; ++e) { af[q][e] = lo[e]; af[q][e + 8] = hi[e]; }
        }
#pragma unroll
        for (int q = 0; q < 2; ++q) {
            const bf16* p = &Bs[cur][(wn + q * 16 + lrow) * LDT + half * 8];
            v8bf lo = *(const v8bf*)p;
            v8bf hi = *(const v8bf*)(p + 16);
#pragma unroll
            for (int e = 0; e < 8; ++e) { bfr[q][e] = lo[e]; bfr[q][e + 8] = hi[e]; }
        }
#pragma unroll
        for (int qi = 0; qi < 4; ++qi)
#pragma unroll
            for (int qj = 0; qj < 2; ++qj)
                acc[qi][qj] = __builtin_amdgcn_wmma_f32_16x16x32_bf16(
                    false, af[qi], false, bfr[qj], (short)0, acc[qi][qj], false, false);
        __syncthreads();   // protects buffer reuse by next stage()
    }

    // epilogue: C frag layout -> row = r + half*8, col = lane&15
#pragma unroll
    for (int qi = 0; qi < 4; ++qi)
#pragma unroll
        for (int qj = 0; qj < 2; ++qj) {
            int gcol = bN + wn + qj * 16 + lrow;
            float bcol = bias[gcol];
#pragma unroll
            for (int r = 0; r < 8; ++r) {
                int grow = bM + wm + qi * 16 + half * 8 + r;
                float c = acc[qi][qj][r] + bcol;
                if constexpr (MODE == 0) {
                    ((float*)Cptr)[(size_t)grow * Ncols + gcol] = c;
                } else if constexpr (MODE == 1) {
                    float g = 0.5f * c * (1.0f + erff(c * 0.70710678118654752f));
                    ((bf16*)Cptr)[(size_t)grow * Ncols + gcol] = (bf16)g;
                } else {
                    float wv = wrow[(size_t)grow * (NG * NE)];
                    float* o = (float*)Cptr + (size_t)grow * Ncols + gcol;
                    *o += c * wv;
                }
            }
        }
}

// ------------------------------------------------- hierarchical router
__global__ __launch_bounds__(256) void router_kernel(
    const float* __restrict__ z, const float* __restrict__ Wg, const float* __restrict__ bg,
    const float* __restrict__ Wr, const float* __restrict__ br, float* __restrict__ wbuf) {
    int wid = threadIdx.x >> 5, lane = threadIdx.x & 31;
    int n = blockIdx.x * 8 + wid;

    float ag[NG] = {0.f, 0.f, 0.f};
    float ae[NG * NE];
#pragma unroll
    for (int i = 0; i < NG * NE; ++i) ae[i] = 0.f;

    for (int h = lane; h < HDIM; h += 32) {
        float zv = z[(size_t)n * HDIM + h];
#pragma unroll
        for (int g = 0; g < NG; ++g) ag[g] += zv * Wg[h * NG + g];
#pragma unroll
        for (int g = 0; g < NG; ++g)
#pragma unroll
            for (int e = 0; e < NE; ++e)
                ae[g * NE + e] += zv * Wr[((size_t)g * HDIM + h) * NE + e];
    }
#pragma unroll
    for (int off = 16; off > 0; off >>= 1) {
#pragma unroll
        for (int g = 0; g < NG; ++g) ag[g] += __shfl_xor(ag[g], off, 32);
#pragma unroll
        for (int i = 0; i < NG * NE; ++i) ae[i] += __shfl_xor(ae[i], off, 32);
    }
    if (lane == 0) {
#pragma unroll
        for (int g = 0; g < NG; ++g) ag[g] += bg[g];
        float m = fmaxf(ag[0], fmaxf(ag[1], ag[2]));
        float eg[NG], s = 0.f;
#pragma unroll
        for (int g = 0; g < NG; ++g) { eg[g] = expf(ag[g] - m); s += eg[g]; }
        float inv = 1.0f / s;
        for (int g = 0; g < NG; ++g) {
            float lm = -3.0e38f;
#pragma unroll
            for (int e = 0; e < NE; ++e) lm = fmaxf(lm, ae[g * NE + e] + br[g * NE + e]);
            float ev[NE], es = 0.f;
#pragma unroll
            for (int e = 0; e < NE; ++e) { ev[e] = expf(ae[g * NE + e] + br[g * NE + e] - lm); es += ev[e]; }
            float pg = eg[g] * inv / es;
#pragma unroll
            for (int e = 0; e < NE; ++e)
                wbuf[(size_t)n * (NG * NE) + g * NE + e] = pg * ev[e];
        }
    }
}

// ---------------------------------------------------------------- driver
extern "C" void kernel_launch(void* const* d_in, const int* in_sizes, int n_in,
                              void* d_out, int out_size, void* d_ws, size_t ws_size,
                              hipStream_t stream) {
    const float* x    = (const float*)d_in[0];
    const float* cond = (const float*)d_in[1];
    const float* Wc   = (const float*)d_in[2];
    const float* bc   = (const float*)d_in[3];
    const float* Wg   = (const float*)d_in[4];
    const float* bg   = (const float*)d_in[5];
    const float* Wr   = (const float*)d_in[6];
    const float* br   = (const float*)d_in[7];
    const float* W1   = (const float*)d_in[8];
    const float* b1   = (const float*)d_in[9];
    const float* W2   = (const float*)d_in[10];
    const float* b2   = (const float*)d_in[11];
    float* out = (float*)d_out;

    char* ws = (char*)d_ws;
    size_t off = 0;
    float* z    = (float*)(ws + off); off += (size_t)N_TOK * HDIM * 4;       // 8 MB
    float* wbuf = (float*)(ws + off); off += (size_t)N_TOK * NG * NE * 4;    // 192 KB
    bf16*  xb   = (bf16*) (ws + off); off += (size_t)N_TOK * HDIM * 2;       // 4 MB
    bf16*  xcb  = (bf16*) (ws + off); off += (size_t)N_TOK * 2 * HDIM * 2;   // 8 MB
    bf16*  WcT  = (bf16*) (ws + off); off += (size_t)HDIM * 2 * HDIM * 2;    // 4 MB
    bf16*  hbuf = (bf16*) (ws + off); off += (size_t)N_TOK * IDIM * 2;       // 16 MB
    bf16*  w1t  = (bf16*) (ws + off); off += (size_t)IDIM * HDIM * 2;        // 8 MB
    bf16*  w2t  = (bf16*) (ws + off); off += (size_t)HDIM * IDIM * 2;        // 8 MB

    // residual base + bf16 packing of activations
    init_out_kernel<<<(N_TOK * HDIM) / (256 * 4), 256, 0, stream>>>(x, out);
    prep_x_kernel<<<(N_TOK * HDIM) / (256 * 4), 256, 0, stream>>>(x, cond, xb, xcb);

    // Wc [2H][H] -> WcT [H][2H];  z = [x|cond]@Wc + bc
    transpose_conv_kernel<<<dim3(HDIM / 32, 2 * HDIM / 32), 256, 0, stream>>>(
        Wc, WcT, 2 * HDIM, HDIM);
    gemm_bf16_async_kernel<0><<<dim3(HDIM / BN, N_TOK / BM), 256, 0, stream>>>(
        xcb, WcT, bc, z, nullptr, 2 * HDIM, HDIM);

    router_kernel<<<N_TOK / 8, 256, 0, stream>>>(z, Wg, bg, Wr, br, wbuf);

    for (int g = 0; g < NG; ++g)
        for (int e = 0; e < NE; ++e) {
            int ge = g * NE + e;
            const float* W1ge = W1 + (size_t)ge * HDIM * IDIM;
            const float* b1ge = b1 + (size_t)ge * IDIM;
            const float* W2ge = W2 + (size_t)ge * IDIM * HDIM;
            const float* b2ge = b2 + (size_t)ge * HDIM;
            // W1 [H][I] -> w1t [I][H]
            transpose_conv_kernel<<<dim3(IDIM / 32, HDIM / 32), 256, 0, stream>>>(
                W1ge, w1t, HDIM, IDIM);
            gemm_bf16_async_kernel<1><<<dim3(IDIM / BN, N_TOK / BM), 256, 0, stream>>>(
                xb, w1t, b1ge, hbuf, nullptr, HDIM, IDIM);
            // W2 [I][H] -> w2t [H][I]
            transpose_conv_kernel<<<dim3(HDIM / 32, IDIM / 32), 256, 0, stream>>>(
                W2ge, w2t, IDIM, HDIM);
            gemm_bf16_async_kernel<2><<<dim3(HDIM / BN, N_TOK / BM), 256, 0, stream>>>(
                hbuf, w2t, b2ge, out, wbuf + ge, IDIM, HDIM);
        }
}
